// Attention2_18571438588360
// MI455X (gfx1250) — matrix operand order repaired and checked by reading, not compiled, not run on hardware
//
#include <hip/hip_runtime.h>
#include <hip/hip_bf16.h>

// MI455X / gfx1250: wave32, WMMA 16x16x32 bf16 + async global->LDS copies.
typedef __attribute__((ext_vector_type(16))) __bf16 v16bf;
typedef __attribute__((ext_vector_type(8)))  float  v8f;

#define LDX 264   // padded row stride (bf16 elems) for 256-wide LDS tiles (528B, 16B-aligned)
#define LDV 40    // padded row stride for 32-wide LDS tiles (80B, 16B-aligned)

// ---- gfx1250 async global->LDS (ASYNCcnt-tracked) -------------------------
// 16-byte per-lane copy: VDST = LDS byte offset (u32 VGPR), VADDR = 64-bit
// global address. Generic pointers to __shared__ carry the LDS offset in their
// low 32 bits on amdgcn.
static __device__ inline void async_ld16(const void* lds_dst, const void* gsrc) {
  unsigned lds_off = (unsigned)(size_t)lds_dst;
  unsigned long long ga = (unsigned long long)(size_t)gsrc;
  asm volatile("global_load_async_to_lds_b128 %0, %1, off"
               :
               : "v"(lds_off), "v"(ga)
               : "memory");
}

#if __has_builtin(__builtin_amdgcn_s_wait_asynccnt)
#define WAIT_ASYNC(n) __builtin_amdgcn_s_wait_asynccnt(n)
#else
#define WAIT_ASYNC(n) asm volatile("s_wait_asynccnt %0" ::"i"(n) : "memory")
#endif

// A-matrix fragment (16x32 bf16): lane L<16 -> row M=L, K in {0..7, 16..23};
// lane L>=16 -> row M=L-16, K in {8..15, 24..31}. Two contiguous 16B runs.
static __device__ inline v16bf load_a_frag(const __bf16* __restrict__ tile,
                                           int row0, int k0, int ld) {
  const int lane = threadIdx.x & 31;
  const int r = lane & 15, hi = lane >> 4;
  const __bf16* p = tile + (row0 + r) * ld + k0;
  v16bf a;
#pragma unroll
  for (int e = 0; e < 8; ++e) a[e] = p[hi * 8 + e];
#pragma unroll
  for (int e = 0; e < 8; ++e) a[8 + e] = p[16 + hi * 8 + e];
  return a;
}

// B-matrix fragment (32x16 bf16) from a tile stored TRANSPOSED as [n][k]:
// lane L<16 -> col N=L, K in {0..7, 16..23}; lane L>=16 -> col N=L-16, K in {8..15, 24..31}.
static __device__ inline v16bf load_bt_frag(const __bf16* __restrict__ tile,
                                            int n0, int k0, int ld) {
  const int lane = threadIdx.x & 31;
  const int n = lane & 15, hi = lane >> 4;
  const __bf16* p = tile + (n0 + n) * ld + k0;
  v16bf b;
#pragma unroll
  for (int e = 0; e < 8; ++e) b[e] = p[hi * 8 + e];
#pragma unroll
  for (int e = 0; e < 8; ++e) b[8 + e] = p[16 + hi * 8 + e];
  return b;
}

static __device__ inline v8f wmma_bf16(v16bf a, v16bf b, v8f c) {
  return __builtin_amdgcn_wmma_f32_16x16x32_bf16(false, a, false, b, (short)0, c,
                                                 false, false);
}

// ---------------------------------------------------------------------------
// Kernel 0: W[d][a] fp32 -> Wt[w][a][d] bf16 (transposed for B-frag loads)
// ---------------------------------------------------------------------------
__global__ void prep_w_kernel(const float* __restrict__ W1,
                              const float* __restrict__ W2,
                              const float* __restrict__ W3,
                              __bf16* __restrict__ Wt) {
  int w = blockIdx.x >> 8;
  int a = blockIdx.x & 255;
  int d = threadIdx.x;
  const float* W = (w == 0) ? W1 : (w == 1) ? W2 : W3;
  Wt[((size_t)w * 256 + a) * 256 + d] = (__bf16)W[d * 256 + a];
}

// ---------------------------------------------------------------------------
// Kernel 1: QKV projection. Q,K stored row-major [tok][a]; V stored
// TRANSPOSED per batch as Vt[b][a][tok] so attention can async-copy it.
// ---------------------------------------------------------------------------
__global__ __launch_bounds__(256) void qkv_kernel(
    const float* __restrict__ x, const __bf16* __restrict__ Wt,
    const float* __restrict__ b1, const float* __restrict__ b2,
    const float* __restrict__ b3, __bf16* __restrict__ Qb,
    __bf16* __restrict__ Kb, __bf16* __restrict__ Vt) {
  __shared__ alignas(16) __bf16 Xs[128 * LDX];
  __shared__ alignas(16) __bf16 Ws[64 * LDX];

  const int tid = threadIdx.x;
  const int mt = blockIdx.y;  // 128-row tile
  const int nt = blockIdx.x;  // 64-col tile
  const int row0 = mt * 128;
  const int col0 = nt * 64;

  for (int t = tid; t < 128 * 256; t += 256) {
    int r = t >> 8, d = t & 255;
    Xs[r * LDX + d] = (__bf16)x[(size_t)(row0 + r) * 256 + d];
  }

  const int wave = tid >> 5;
  const int wr = wave >> 1;
  const int wc = wave & 1;
  const int m0 = wr * 32;
  const int n0 = wc * 32;
  const int lane = tid & 31;
  const int cn = lane & 15, chi = lane >> 4;

  for (int w = 0; w < 3; ++w) {
    __syncthreads();
    const __bf16* wsrc = Wt + ((size_t)w * 256 + col0) * 256;
    for (int t = tid; t < 64 * 128; t += 256) {
      int a = t >> 7, dw = t & 127;
      ((unsigned int*)(Ws + a * LDX))[dw] =
          ((const unsigned int*)(wsrc + (size_t)a * 256))[dw];
    }
    __syncthreads();

    const float* bias = (w == 0) ? b1 : (w == 1) ? b2 : b3;
    v8f c[2][2];
#pragma unroll
    for (int i = 0; i < 2; ++i)
#pragma unroll
      for (int j = 0; j < 2; ++j) {
        float bv = bias[col0 + n0 + j * 16 + cn];
#pragma unroll
        for (int e = 0; e < 8; ++e) c[i][j][e] = bv;
      }

#pragma unroll
    for (int kb = 0; kb < 8; ++kb) {
      v16bf a0 = load_a_frag(Xs, m0, kb * 32, LDX);
      v16bf a1 = load_a_frag(Xs, m0 + 16, kb * 32, LDX);
      v16bf bb0 = load_bt_frag(Ws, n0, kb * 32, LDX);
      v16bf bb1 = load_bt_frag(Ws, n0 + 16, kb * 32, LDX);
      c[0][0] = wmma_bf16(a0, bb0, c[0][0]);
      c[0][1] = wmma_bf16(a0, bb1, c[0][1]);
      c[1][0] = wmma_bf16(a1, bb0, c[1][0]);
      c[1][1] = wmma_bf16(a1, bb1, c[1][1]);
    }

#pragma unroll
    for (int i = 0; i < 2; ++i)
#pragma unroll
      for (int j = 0; j < 2; ++j)
#pragma unroll
        for (int e = 0; e < 8; ++e) {
          int grow = row0 + m0 + i * 16 + chi * 8 + e;
          int gcol = col0 + n0 + j * 16 + cn;
          if (w == 0) {
            Qb[(size_t)grow * 256 + gcol] = (__bf16)c[i][j][e];
          } else if (w == 1) {
            Kb[(size_t)grow * 256 + gcol] = (__bf16)c[i][j][e];
          } else {
            int bb2 = grow >> 11, tok = grow & 2047;  // batch, token-in-batch
            Vt[((size_t)bb2 * 256 + gcol) * 2048 + tok] = (__bf16)c[i][j][e];
          }
        }
  }
}

// ---------------------------------------------------------------------------
// Kernel 2: flash attention, async double-buffered K/V staging.
// Grid (qt=16, b=16). 8 waves; each wave owns 16 queries x full 256-wide ctx.
// ---------------------------------------------------------------------------
__global__ __launch_bounds__(256) void attn_kernel(
    const __bf16* __restrict__ Qb, const __bf16* __restrict__ Kb,
    const __bf16* __restrict__ Vt, float* __restrict__ partial) {
  __shared__ alignas(16) __bf16 Qs[128 * LDX];       // 128 q x 256
  __shared__ alignas(16) __bf16 Ks[2][32 * LDX];     // double-buffered keys
  __shared__ alignas(16) __bf16 Vts[2][256 * LDV];   // double-buffered V^T
  __shared__ alignas(16) __bf16 Ps[8][16 * LDV];     // per-wave P staging

  const int tid = threadIdx.x;
  const int b = blockIdx.y;
  const int qt = blockIdx.x;
  const size_t base = (size_t)b * 2048 * 256;
  const size_t vtb = (size_t)b * 256 * 2048;
  const int q0 = qt * 128;

  // Stage one 32-key block (K row-major, V^T) into buffer bb: 8 async ops/thread.
  auto stage = [&](int bb, int kblk) {
    const size_t kbase = base + (size_t)kblk * 32 * 256;
#pragma unroll
    for (int c = 0; c < 4; ++c) {
      int chunk = c * 256 + tid;
      int r = chunk >> 5, c8 = (chunk & 31) * 8;  // 32 x16B chunks per row
      async_ld16(&Ks[bb][r * LDX + c8], Kb + kbase + (size_t)r * 256 + c8);
    }
    const size_t vbase = vtb + (size_t)kblk * 32;
#pragma unroll
    for (int c = 0; c < 4; ++c) {
      int chunk = c * 256 + tid;
      int a = chunk >> 2, c8 = (chunk & 3) * 8;   // 4 x16B chunks per a-row
      async_ld16(&Vts[bb][a * LDV + c8], Vt + vbase + (size_t)a * 2048 + c8);
    }
  };

  // Prologue: Q tile (16 ops/thread) + key-block 0 (8 ops/thread), all async.
#pragma unroll
  for (int c = 0; c < 16; ++c) {
    int chunk = c * 256 + tid;
    int r = chunk >> 5, c8 = (chunk & 31) * 8;
    async_ld16(Qs + r * LDX + c8, Qb + base + (size_t)(q0 + r) * 256 + c8);
  }
  stage(0, 0);

  const int wave = tid >> 5;
  const int lane = tid & 31;
  const int cn = lane & 15, chi = lane >> 4;

  v8f acc[16];
  float mrow[8], lrow[8];
#pragma unroll
  for (int f = 0; f < 16; ++f)
#pragma unroll
    for (int e = 0; e < 8; ++e) acc[f][e] = 0.f;
#pragma unroll
  for (int i = 0; i < 8; ++i) { mrow[i] = -1e30f; lrow[i] = 0.f; }

  for (int kb = 0; kb < 64; ++kb) {
    const int cur = kb & 1;
    if (kb + 1 < 64) {
      stage(cur ^ 1, kb + 1);  // overlap next block with the wait below
      WAIT_ASYNC(8);           // everything through block kb has landed
    } else {
      WAIT_ASYNC(0);
    }
    __syncthreads();  // publish this block's LDS across all waves

    const __bf16* ksb = &Ks[cur][0];
    const __bf16* vsb = &Vts[cur][0];

    // S = Q_wave(16x256) . K^T -> 16x32  (16 WMMAs)
    v8f s0, s1;
#pragma unroll
    for (int e = 0; e < 8; ++e) { s0[e] = 0.f; s1[e] = 0.f; }
#pragma unroll
    for (int dk = 0; dk < 8; ++dk) {
      v16bf aq = load_a_frag(Qs, wave * 16, dk * 32, LDX);
      v16bf bk0 = load_bt_frag(ksb, 0, dk * 32, LDX);
      v16bf bk1 = load_bt_frag(ksb, 16, dk * 32, LDX);
      s0 = wmma_bf16(aq, bk0, s0);
      s1 = wmma_bf16(aq, bk1, s1);
    }

    // Online softmax; row m = chi*8+i lives across the 16 lanes of this half.
    float rmax[8], p0[8], p1[8], rsum[8], scl[8];
#pragma unroll
    for (int i = 0; i < 8; ++i) {
      rmax[i] = fmaxf(s0[i], s1[i]);
#pragma unroll
      for (int msk = 1; msk <= 8; msk <<= 1)
        rmax[i] = fmaxf(rmax[i], __shfl_xor(rmax[i], msk, 32));
    }
#pragma unroll
    for (int i = 0; i < 8; ++i) {
      float mnew = fmaxf(mrow[i], rmax[i]);
      scl[i] = __expf(mrow[i] - mnew);
      p0[i] = __expf(s0[i] - mnew);
      p1[i] = __expf(s1[i] - mnew);
      rsum[i] = p0[i] + p1[i];
      mrow[i] = mnew;
    }
#pragma unroll
    for (int i = 0; i < 8; ++i) {
#pragma unroll
      for (int msk = 1; msk <= 8; msk <<= 1)
        rsum[i] += __shfl_xor(rsum[i], msk, 32);
      lrow[i] = lrow[i] * scl[i] + rsum[i];
    }
#pragma unroll
    for (int f = 0; f < 16; ++f)
#pragma unroll
      for (int i = 0; i < 8; ++i) acc[f][i] *= scl[i];

    // P: C-layout -> A-layout via wave-private LDS.
    __bf16* pw = &Ps[wave][0];
#pragma unroll
    for (int i = 0; i < 8; ++i) {
      pw[(chi * 8 + i) * LDV + cn] = (__bf16)p0[i];
      pw[(chi * 8 + i) * LDV + 16 + cn] = (__bf16)p1[i];
    }
    v16bf ap = load_a_frag(pw, 0, 0, LDV);

    // ctx += P(16x32) . V(32x256)  (16 WMMAs)
#pragma unroll
    for (int f = 0; f < 16; ++f) {
      v16bf bv = load_bt_frag(vsb, f * 16, 0, LDV);
      acc[f] = wmma_bf16(ap, bv, acc[f]);
    }
    __syncthreads();  // all waves done with buffer `cur` before re-staging it
  }

  // Column sums of (acc / l) over this wave's 16 query rows.
#pragma unroll
  for (int f = 0; f < 16; ++f) {
    float cs = 0.f;
#pragma unroll
    for (int i = 0; i < 8; ++i) cs += acc[f][i] / lrow[i];
    cs += __shfl_xor(cs, 16, 32);
    if (lane < 16)
      partial[(size_t)((b * 16 + qt) * 8 + wave) * 256 + f * 16 + cn] = cs;
  }
}

// ---------------------------------------------------------------------------
// Kernel 3: deterministic reduction of 128 partials per batch -> out[B][A]
// ---------------------------------------------------------------------------
__global__ void reduce_kernel(const float* __restrict__ partial,
                              float* __restrict__ out) {
  int b = blockIdx.x, a = threadIdx.x;
  float s = 0.f;
  for (int j = 0; j < 128; ++j) s += partial[(size_t)(b * 128 + j) * 256 + a];
  out[b * 256 + a] = s;
}

extern "C" void kernel_launch(void* const* d_in, const int* in_sizes, int n_in,
                              void* d_out, int out_size, void* d_ws,
                              size_t ws_size, hipStream_t stream) {
  (void)in_sizes; (void)n_in; (void)out_size; (void)ws_size;
  const float* x  = (const float*)d_in[0];
  const float* W1 = (const float*)d_in[1];
  const float* b1 = (const float*)d_in[2];
  const float* W2 = (const float*)d_in[3];
  const float* b2 = (const float*)d_in[4];
  const float* W3 = (const float*)d_in[5];
  const float* b3 = (const float*)d_in[6];
  float* out = (float*)d_out;

  char* ws = (char*)d_ws;
  __bf16* Qb = (__bf16*)ws;  ws += (size_t)32768 * 256 * 2;
  __bf16* Kb = (__bf16*)ws;  ws += (size_t)32768 * 256 * 2;
  __bf16* Vt = (__bf16*)ws;  ws += (size_t)32768 * 256 * 2;  // [b][a][tok]
  __bf16* Wt = (__bf16*)ws;  ws += (size_t)3 * 256 * 256 * 2;
  float* partial = (float*)ws;  // 2048 * 256 floats

  prep_w_kernel<<<768, 256, 0, stream>>>(W1, W2, W3, Wt);
  qkv_kernel<<<dim3(4, 256), 256, 0, stream>>>(x, Wt, b1, b2, b3, Qb, Kb, Vt);
  attn_kernel<<<dim3(16, 16), 256, 0, stream>>>(Qb, Kb, Vt, partial);
  reduce_kernel<<<16, 256, 0, stream>>>(partial, out);
}
